// GroupQueryAttention_58093727645711
// MI455X (gfx1250) — matrix operand order repaired and checked by reading, not compiled, not run on hardware
//
#include <hip/hip_runtime.h>
#include <cstdint>
#include <cstddef>

// ---------------------------------------------------------------- types
typedef __attribute__((ext_vector_type(16))) __bf16       v16bf;
typedef __attribute__((ext_vector_type(8)))  float        v8f;
typedef __attribute__((ext_vector_type(4)))  unsigned int u32x4;
typedef __attribute__((ext_vector_type(4)))  unsigned int tdm_g0_t;
typedef __attribute__((ext_vector_type(8)))  int          tdm_g1_t;
typedef __attribute__((ext_vector_type(4)))  int          tdm_g2_t;
typedef __attribute__((ext_vector_type(8)))  int          tdm_g4_t;

#if defined(__has_builtin)
#  if __has_builtin(__builtin_amdgcn_tensor_load_to_lds)
#    define HAVE_TDM 1
#  endif
#  if __has_builtin(__builtin_amdgcn_s_wait_tensorcnt)
#    define HAVE_TCNT_BUILTIN 1
#  endif
#endif

union Frag {
  u32x4 q[2];
  v16bf v;
};

__device__ __forceinline__ unsigned short f2bf(float f) {
  unsigned int u = __float_as_uint(f);
  u += 0x7FFFu + ((u >> 16) & 1u);   // round-to-nearest-even
  return (unsigned short)(u >> 16);
}

__device__ __forceinline__ v8f wmma_bf16(const Frag& a, const Frag& b, v8f c) {
  return __builtin_amdgcn_wmma_f32_16x16x32_bf16(
      false, a.v, false, b.v, (short)0, c, false, false);
}

// generic shared pointer -> LDS byte offset (addr[31:0] of LDS aperture)
__device__ __forceinline__ unsigned lds_off(const void* p) {
  return (unsigned)(uintptr_t)p;
}

// async global->LDS copy, 16B per lane (ASYNCcnt-tracked)
__device__ __forceinline__ void async_g2l_b128(void* lds, const void* g) {
  asm volatile("global_load_async_to_lds_b128 %0, %1, off"
               :: "v"(lds_off(lds)), "v"(g) : "memory");
}
__device__ __forceinline__ void wait_async0() {
  asm volatile("s_wait_asynccnt 0" ::: "memory");
}

#if defined(HAVE_TDM)
// One 2D-tile DMA: rows x 32 bf16 elems, row stride K elems in memory,
// LDS rows padded to 40 elems (pad 4 DWORDs after every 16 DWORDs).
__device__ __forceinline__ void tdm_load_tile(const unsigned short* gbase,
                                              unsigned ldsbyte,
                                              unsigned K, unsigned rows) {
  unsigned long long ga = (unsigned long long)(uintptr_t)gbase;
  tdm_g0_t g0;
  g0[0] = 1u;                                            // count=1
  g0[1] = ldsbyte;                                       // lds_addr
  g0[2] = (unsigned)(ga & 0xFFFFFFFFull);                // global_addr lo
  g0[3] = (unsigned)((ga >> 32) & 0x01FFFFFFull) | (2u << 30); // hi | type=2
  tdm_g1_t g1;
  g1[0] = (int)((1u << 16)      // data_size = 2B
              | (1u << 20)      // pad_enable
              | (3u << 22)      // pad_interval: 16 DWORDs (one 64B row)
              | (3u << 25));    // pad_amount:   4 DWORDs (16B -> stride 40)
  g1[1] = (int)((K & 0xFFFFu) << 16);                    // tensor_dim0 lo
  g1[2] = (int)((K >> 16) | ((rows & 0xFFFFu) << 16));   // dim0 hi | dim1 lo
  g1[3] = (int)((rows >> 16) | (32u << 16));             // dim1 hi | tile_dim0=32
  g1[4] = (int)(rows & 0xFFFFu);                         // tile_dim1 | tile_dim2=0
  g1[5] = (int)K;                                        // dim0_stride lo
  g1[6] = 0;                                             // dim0_stride hi | dim1_stride lo
  g1[7] = 0;
  tdm_g2_t z4 = {0, 0, 0, 0};
  tdm_g4_t z8 = {0, 0, 0, 0, 0, 0, 0, 0};
  __builtin_amdgcn_tensor_load_to_lds(g0, g1, z4, z4, z8, 0);
}
__device__ __forceinline__ void wait_tensor(void) {
#if defined(HAVE_TCNT_BUILTIN)
  __builtin_amdgcn_s_wait_tensorcnt((short)0);
#else
  asm volatile("s_wait_tensorcnt 0" ::: "memory");
#endif
}
__device__ __forceinline__ void wait_tensor2(void) {
#if defined(HAVE_TCNT_BUILTIN)
  __builtin_amdgcn_s_wait_tensorcnt((short)2);
#else
  asm volatile("s_wait_tensorcnt 2" ::: "memory");
#endif
}
#endif

static constexpr int B_  = 2;
static constexpr int T_  = 2048;
static constexpr int DIN = 2048;
static constexpr int H_  = 32;
static constexpr int HD_ = 64;
static constexpr int KV_ = 8;
static constexpr int M_  = B_ * T_;   // 4096 rows

// ---------------------------------------------------------------- cast
__global__ void cast_f32_to_bf16(const float* __restrict__ s,
                                 unsigned short* __restrict__ d, int n) {
  int i = blockIdx.x * blockDim.x + threadIdx.x;
  if (i < n) d[i] = f2bf(s[i]);
}

// ---------------------------------------------------------------- GEMM
// C[M,N] (f32) = A[M,K] (bf16) * W[N,K]^T (bf16) + bias[N]
// Double-buffered LDS tiles filled by the Tensor Data Mover (wave 0 issues,
// TENSORcnt-synced) so global->LDS DMA overlaps WMMA on the other buffer.
#define BM 128
#define BN 128
#define BK 32
#define LDT 40   // padded LDS row stride (bf16 elems), 80B, 16B-aligned

__global__ __launch_bounds__(256, 2) void gemm_bf16_nt(
    const unsigned short* __restrict__ A,
    const unsigned short* __restrict__ W,
    const float* __restrict__ bias,
    float* __restrict__ C,
    int M, int N, int K)
{
  __shared__ __align__(16) unsigned short As[2][BM * LDT];
  __shared__ __align__(16) unsigned short Bs[2][BN * LDT];

  const int tid  = threadIdx.x;
  const int lane = tid & 31;
  const int wave = tid >> 5;
  const int wr   = wave & 3;   // 4 wave-rows of 32
  const int wc   = wave >> 2;  // 2 wave-cols of 64
  const int m0   = blockIdx.y * BM;
  const int n0   = blockIdx.x * BN;

  v8f acc[2][4];
#pragma unroll
  for (int i = 0; i < 2; ++i)
#pragma unroll
    for (int j = 0; j < 4; ++j) acc[i][j] = {};

#if defined(HAVE_TDM)
  const unsigned short* gA = A + (size_t)m0 * K;
  const unsigned short* gW = W + (size_t)n0 * K;
  const unsigned la[2] = { lds_off(&As[0][0]), lds_off(&As[1][0]) };
  const unsigned lb[2] = { lds_off(&Bs[0][0]), lds_off(&Bs[1][0]) };
  if (wave == 0) {
    tdm_load_tile(gA, la[0], (unsigned)K, BM);
    tdm_load_tile(gW, lb[0], (unsigned)K, BN);
  }
#else
  const int lrow = tid >> 1;           // 0..127
  const int lkp  = (tid & 1) * 16;     // 0 or 16
  const unsigned short* ga = A + (size_t)(m0 + lrow) * K + lkp;
  const unsigned short* gw = W + (size_t)(n0 + lrow) * K + lkp;
#endif

  int cur = 0;
  for (int kk = 0; kk < K; kk += BK) {
    const bool more = (kk + BK) < K;
#if defined(HAVE_TDM)
    if (wave == 0) {
      if (more) {                       // DMA next tiles into other buffer
        tdm_load_tile(gA + kk + BK, la[cur ^ 1], (unsigned)K, BM);
        tdm_load_tile(gW + kk + BK, lb[cur ^ 1], (unsigned)K, BN);
        wait_tensor2();                 // current buffer's 2 DMAs done
      } else {
        wait_tensor();
      }
    }
    __syncthreads();
#else
    {
      u32x4 a0 = *(const u32x4*)(ga + kk);
      u32x4 a1 = *(const u32x4*)(ga + kk + 8);
      u32x4 b0 = *(const u32x4*)(gw + kk);
      u32x4 b1 = *(const u32x4*)(gw + kk + 8);
      *(u32x4*)(&As[cur][lrow * LDT + lkp])     = a0;
      *(u32x4*)(&As[cur][lrow * LDT + lkp + 8]) = a1;
      *(u32x4*)(&Bs[cur][lrow * LDT + lkp])     = b0;
      *(u32x4*)(&Bs[cur][lrow * LDT + lkp + 8]) = b1;
      if (more) {
        __builtin_prefetch(ga + kk + BK, 0, 1);
        __builtin_prefetch(gw + kk + BK, 0, 1);
      }
    }
    __syncthreads();
#endif

    // A fragments: 16x32 bf16, row = lane%16, K chunks split per half-wave
    Frag afr[2];
    const int ar  = wr * 32 + (lane & 15);
    const int ak0 = (lane < 16) ? 0 : 8;
#pragma unroll
    for (int mt = 0; mt < 2; ++mt) {
      const int r = ar + mt * 16;
      afr[mt].q[0] = *(const u32x4*)(&As[cur][r * LDT + ak0]);
      afr[mt].q[1] = *(const u32x4*)(&As[cur][r * LDT + 16 + ak0]);
    }
    // B fragments: 32x16 bf16, col = lane%16, K chunks split per half-wave
    const int bk0 = (lane < 16) ? 0 : 8;
#pragma unroll
    for (int nt = 0; nt < 4; ++nt) {
      const int c = wc * 64 + nt * 16 + (lane & 15);
      Frag bfr;
      bfr.q[0] = *(const u32x4*)(&Bs[cur][c * LDT + bk0]);
      bfr.q[1] = *(const u32x4*)(&Bs[cur][c * LDT + bk0 + 16]);
#pragma unroll
      for (int mt = 0; mt < 2; ++mt)
        acc[mt][nt] = wmma_bf16(afr[mt], bfr, acc[mt][nt]);
    }
    __syncthreads();   // all waves done reading before buffer is re-filled
    cur ^= 1;
  }

  // epilogue: C/D layout -> VGPR v = rows v / v+8, lane%16 = col
#pragma unroll
  for (int mt = 0; mt < 2; ++mt)
#pragma unroll
    for (int nt = 0; nt < 4; ++nt) {
      const int gn = n0 + wc * 64 + nt * 16 + (lane & 15);
      const float bvv = bias[gn];
#pragma unroll
      for (int v = 0; v < 8; ++v) {
        const int gm = m0 + wr * 32 + mt * 16 + v + ((lane >= 16) ? 8 : 0);
        C[(size_t)gm * N + gn] = acc[mt][nt][v] + bvv;
      }
    }
}

// ---------------------------------------------------------------- RoPE / layout
__global__ void rope_q_kernel(const float* __restrict__ qf,
                              const float* __restrict__ cosb,
                              const float* __restrict__ sinb,
                              unsigned short* __restrict__ qb, int n) {
  int i = blockIdx.x * blockDim.x + threadIdx.x;
  if (i >= n) return;
  const int p = i & 31;
  const int h = (i >> 5) & (H_ - 1);
  const int t = (i >> 10) & (T_ - 1);
  const int b = i >> 21;
  const float* src = qf + (size_t)(b * T_ + t) * (H_ * HD_) + h * HD_ + 2 * p;
  const float x1 = src[0], x2 = src[1];
  const float cc = cosb[t * 32 + p], ss = sinb[t * 32 + p];
  unsigned short* dst = qb + ((size_t)(b * H_ + h) * T_ + t) * HD_ + 2 * p;
  dst[0] = f2bf(x1 * cc - x2 * ss);
  dst[1] = f2bf(x1 * ss + x2 * cc);
}

__global__ void rope_k_kernel(const float* __restrict__ kf,
                              const float* __restrict__ cosb,
                              const float* __restrict__ sinb,
                              unsigned short* __restrict__ kb,
                              float* __restrict__ kout, int n) {
  int i = blockIdx.x * blockDim.x + threadIdx.x;
  if (i >= n) return;
  const int p  = i & 31;
  const int kv = (i >> 5) & (KV_ - 1);
  const int t  = (i >> 8) & (T_ - 1);
  const int b  = i >> 19;
  const float* src = kf + (size_t)(b * T_ + t) * (KV_ * HD_) + kv * HD_ + 2 * p;
  const float x1 = src[0], x2 = src[1];
  const float cc = cosb[t * 32 + p], ss = sinb[t * 32 + p];
  const float e = x1 * cc - x2 * ss;
  const float o = x1 * ss + x2 * cc;
  const size_t di = ((size_t)(b * KV_ + kv) * T_ + t) * HD_ + 2 * p;
  kout[di] = e;  kout[di + 1] = o;            // f32 tuple output (b,kv,t,hd)
  kb[di]   = f2bf(e);  kb[di + 1] = f2bf(o);  // bf16 for attention
}

__global__ void v_post_kernel(const float* __restrict__ vf,
                              unsigned short* __restrict__ vb,
                              float* __restrict__ vout, int n) {
  int i = blockIdx.x * blockDim.x + threadIdx.x;
  if (i >= n) return;
  const int d  = i & (HD_ - 1);
  const int kv = (i >> 6) & (KV_ - 1);
  const int t  = (i >> 9) & (T_ - 1);
  const int b  = i >> 20;
  const float val = vf[(size_t)(b * T_ + t) * (KV_ * HD_) + kv * HD_ + d];
  const size_t di = ((size_t)(b * KV_ + kv) * T_ + t) * HD_ + d;
  vout[di] = val;
  vb[di]   = f2bf(val);
}

// ---------------------------------------------------------------- flash GQA
// grid (T/64, B*H), 128 threads = 4 waves; each wave owns 16 q-rows.
// K chunks arrive via async global->LDS (ASYNCcnt); V transposed via regs.
__global__ __launch_bounds__(128, 2) void flash_gqa(
    const unsigned short* __restrict__ qb,
    const unsigned short* __restrict__ kb,
    const unsigned short* __restrict__ vb,
    unsigned short* __restrict__ ctx)
{
  __shared__ __align__(16) unsigned short Ks[64 * 72];      // K chunk [t][hd]
  __shared__ __align__(16) unsigned short Vt[64 * 72];      // V chunk [hd][t]
  __shared__ __align__(16) unsigned short Ps[4][16 * 72];   // per-wave P tile

  const int tid  = threadIdx.x;
  const int lane = tid & 31;
  const int wave = tid >> 5;
  const int qt   = blockIdx.x;
  const int bh   = blockIdx.y;
  const int h    = bh & (H_ - 1);
  const int b    = bh >> 5;
  const int kvh  = h >> 2;                   // group size 4
  const int qbase = qt * 64 + wave * 16;

  // persistent Q fragments (16x64 -> two 16x32 A-frags)
  const unsigned short* qr =
      qb + ((size_t)(b * H_ + h) * T_ + qbase + (lane & 15)) * HD_;
  const int ak0 = (lane < 16) ? 0 : 8;
  Frag aq[2];
  aq[0].q[0] = *(const u32x4*)(qr + ak0);
  aq[0].q[1] = *(const u32x4*)(qr + 16 + ak0);
  aq[1].q[0] = *(const u32x4*)(qr + 32 + ak0);
  aq[1].q[1] = *(const u32x4*)(qr + 48 + ak0);

  v8f o[4];
#pragma unroll
  for (int nt = 0; nt < 4; ++nt) o[nt] = {};
  float mi[8], li[8];
#pragma unroll
  for (int v = 0; v < 8; ++v) { mi[v] = -1e30f; li[v] = 0.0f; }

  const int lrow = tid >> 1;
  const int cp   = (tid & 1) * 32;
  const size_t kvbase = (size_t)(b * KV_ + kvh) * T_;

  for (int c = 0; c <= qt; ++c) {
    const int kbase = c * 64;
    const unsigned short* kg = kb + (kvbase + kbase + lrow) * HD_ + cp;
    const unsigned short* vg = vb + (kvbase + kbase + lrow) * HD_ + cp;
    // K: async DMA straight into LDS, no VGPR round-trip
    async_g2l_b128(&Ks[lrow * 72 + cp],     kg);
    async_g2l_b128(&Ks[lrow * 72 + cp + 8], kg + 8);
    // V: load to regs, scatter transposed into Vt
    union { u32x4 q[4]; unsigned short u[32]; } vr;
#pragma unroll
    for (int j = 0; j < 4; ++j) vr.q[j] = *(const u32x4*)(vg + j * 8);
#pragma unroll
    for (int j = 0; j < 32; ++j) Vt[(cp + j) * 72 + lrow] = vr.u[j];
    wait_async0();
    __syncthreads();

    const bool diag = (c == qt);
    const int  radd = (lane >= 16) ? 8 : 0;

    for (int jt = 0; jt < 4; ++jt) {
      // S(16x16) = Q(16x64) * K_tile^T : two k=32 WMMAs
      v8f s = {};
      const int jl = jt * 16 + (lane & 15);
      const int d0 = (lane < 16) ? 0 : 8;
      Frag bk0, bk1;
      bk0.q[0] = *(const u32x4*)(&Ks[jl * 72 + d0]);
      bk0.q[1] = *(const u32x4*)(&Ks[jl * 72 + d0 + 16]);
      bk1.q[0] = *(const u32x4*)(&Ks[jl * 72 + 32 + d0]);
      bk1.q[1] = *(const u32x4*)(&Ks[jl * 72 + 32 + d0 + 16]);
      s = wmma_bf16(aq[0], bk0, s);
      s = wmma_bf16(aq[1], bk1, s);

      const int gcol = kbase + jt * 16 + (lane & 15);
#pragma unroll
      for (int v = 0; v < 8; ++v) {
        float sv = s[v] * 0.125f;                       // 1/sqrt(64)
        const int grow = qbase + v + radd;
        if (diag && gcol > grow) sv = -1e30f;           // causal mask
        // row-max over the 16 lanes of this half-wave
        float rm = sv;
        rm = fmaxf(rm, __shfl_xor(rm, 1));
        rm = fmaxf(rm, __shfl_xor(rm, 2));
        rm = fmaxf(rm, __shfl_xor(rm, 4));
        rm = fmaxf(rm, __shfl_xor(rm, 8));
        const float mn    = fmaxf(mi[v], rm);
        const float alpha = __expf(mi[v] - mn);
        float p = (sv > -1e29f) ? __expf(sv - mn) : 0.0f;
        float ps = p;
        ps += __shfl_xor(ps, 1);
        ps += __shfl_xor(ps, 2);
        ps += __shfl_xor(ps, 4);
        ps += __shfl_xor(ps, 8);
        li[v] = li[v] * alpha + ps;
        mi[v] = mn;
#pragma unroll
        for (int nt = 0; nt < 4; ++nt) o[nt][v] *= alpha;
        Ps[wave][(v + radd) * 72 + jt * 16 + (lane & 15)] = f2bf(p);
      }
    }

    // O(16x64) += P(16x64) * V(64x64) : 2 halves x 4 n-tiles
#pragma unroll
    for (int half = 0; half < 2; ++half) {
      Frag ap;
      const int pr  = lane & 15;
      const int pk0 = half * 32 + ((lane < 16) ? 0 : 8);
      ap.q[0] = *(const u32x4*)(&Ps[wave][pr * 72 + pk0]);
      ap.q[1] = *(const u32x4*)(&Ps[wave][pr * 72 + pk0 + 16]);
      const int kv0 = half * 32 + ((lane < 16) ? 0 : 8);
#pragma unroll
      for (int nt = 0; nt < 4; ++nt) {
        const int n = nt * 16 + (lane & 15);
        Frag bvf;
        bvf.q[0] = *(const u32x4*)(&Vt[n * 72 + kv0]);
        bvf.q[1] = *(const u32x4*)(&Vt[n * 72 + kv0 + 16]);
        o[nt] = wmma_bf16(ap, bvf, o[nt]);
      }
    }
    __syncthreads();
  }

  // normalize + write ctx as bf16 [b][t][h*hd]
  const int radd = (lane >= 16) ? 8 : 0;
#pragma unroll
  for (int v = 0; v < 8; ++v) {
    const float inv = 1.0f / li[v];
    const int trow = qbase + v + radd;
#pragma unroll
    for (int nt = 0; nt < 4; ++nt)
      ctx[((size_t)b * T_ + trow) * (H_ * HD_) + h * HD_ + nt * 16 + (lane & 15)] =
          f2bf(o[nt][v] * inv);
  }
}

// ---------------------------------------------------------------- launch
extern "C" void kernel_launch(void* const* d_in, const int* in_sizes, int n_in,
                              void* d_out, int out_size, void* d_ws, size_t ws_size,
                              hipStream_t stream) {
  (void)in_sizes; (void)n_in; (void)out_size; (void)ws_size;
  const float* x    = (const float*)d_in[0];
  const float* cosb = (const float*)d_in[1];
  const float* sinb = (const float*)d_in[2];
  const float* wq   = (const float*)d_in[3];
  const float* bq   = (const float*)d_in[4];
  const float* wk   = (const float*)d_in[5];
  const float* bk   = (const float*)d_in[6];
  const float* wv   = (const float*)d_in[7];
  const float* bv   = (const float*)d_in[8];
  const float* wo   = (const float*)d_in[9];
  const float* bo   = (const float*)d_in[10];

  float* out  = (float*)d_out;                         // (b,t,2048)
  float* kout = out + (size_t)B_ * T_ * DIN;           // (b,kv,t,hd)
  float* vout = kout + (size_t)B_ * KV_ * T_ * HD_;    // (b,kv,t,hd)

  char* w = (char*)d_ws;
  auto take = [&](size_t bytes) {
    char* p = w; w += (bytes + 255) & ~(size_t)255; return p;
  };
  unsigned short* x_bf  = (unsigned short*)take((size_t)M_ * DIN * 2);
  unsigned short* wq_bf = (unsigned short*)take((size_t)H_ * HD_ * DIN * 2);
  unsigned short* wk_bf = (unsigned short*)take((size_t)KV_ * HD_ * DIN * 2);
  unsigned short* wv_bf = (unsigned short*)take((size_t)KV_ * HD_ * DIN * 2);
  unsigned short* wo_bf = (unsigned short*)take((size_t)DIN * H_ * HD_ * 2);
  float* q_f = (float*)take((size_t)M_ * H_ * HD_ * 4);
  float* k_f = (float*)take((size_t)M_ * KV_ * HD_ * 4);
  float* v_f = (float*)take((size_t)M_ * KV_ * HD_ * 4);
  unsigned short* q_bf = (unsigned short*)take((size_t)M_ * H_ * HD_ * 2);
  unsigned short* k_bf = (unsigned short*)take((size_t)M_ * KV_ * HD_ * 2);
  unsigned short* v_bf = (unsigned short*)take((size_t)M_ * KV_ * HD_ * 2);
  unsigned short* ctx_bf = (unsigned short*)q_f;  // q_f dead after RoPE; reuse

  auto cdiv = [](int a, int b) { return (a + b - 1) / b; };

  { int n = M_ * DIN;        cast_f32_to_bf16<<<cdiv(n,256),256,0,stream>>>(x,  x_bf,  n); }
  { int n = H_ * HD_ * DIN;  cast_f32_to_bf16<<<cdiv(n,256),256,0,stream>>>(wq, wq_bf, n); }
  { int n = KV_ * HD_ * DIN; cast_f32_to_bf16<<<cdiv(n,256),256,0,stream>>>(wk, wk_bf, n); }
  { int n = KV_ * HD_ * DIN; cast_f32_to_bf16<<<cdiv(n,256),256,0,stream>>>(wv, wv_bf, n); }
  { int n = DIN * H_ * HD_;  cast_f32_to_bf16<<<cdiv(n,256),256,0,stream>>>(wo, wo_bf, n); }

  gemm_bf16_nt<<<dim3((H_*HD_)/BN,  M_/BM), 256, 0, stream>>>(x_bf, wq_bf, bq, q_f, M_, H_*HD_,  DIN);
  gemm_bf16_nt<<<dim3((KV_*HD_)/BN, M_/BM), 256, 0, stream>>>(x_bf, wk_bf, bk, k_f, M_, KV_*HD_, DIN);
  gemm_bf16_nt<<<dim3((KV_*HD_)/BN, M_/BM), 256, 0, stream>>>(x_bf, wv_bf, bv, v_f, M_, KV_*HD_, DIN);

  { int n = B_*T_*H_*(HD_/2);  rope_q_kernel<<<cdiv(n,256),256,0,stream>>>(q_f, cosb, sinb, q_bf, n); }
  { int n = B_*T_*KV_*(HD_/2); rope_k_kernel<<<cdiv(n,256),256,0,stream>>>(k_f, cosb, sinb, k_bf, kout, n); }
  { int n = B_*T_*KV_*HD_;     v_post_kernel<<<cdiv(n,256),256,0,stream>>>(v_f, v_bf, vout, n); }

  flash_gqa<<<dim3(T_/64, B_*H_), 128, 0, stream>>>(q_bf, k_bf, v_bf, ctx_bf);

  gemm_bf16_nt<<<dim3(DIN/BN, M_/BM), 256, 0, stream>>>(ctx_bf, wo_bf, bo, out, M_, DIN, DIN);
}
